// LocalGlobalBlock_43439299232202
// MI455X (gfx1250) — matrix operand order repaired and checked by reading, not compiled, not run on hardware
//
#include <hip/hip_runtime.h>
#include <math.h>

#define B_   16
#define MID  128
#define CIN2 256          // 2*MID
#define OUTC 512
#define H_   112
#define W_   112
#define HW   (H_*W_)      // 12544
#define OH   56
#define OW   56
#define OHW  (OH*OW)      // 3136
#define KDS  (9*CIN2)     // 2304
#define EPSV 1e-5f

#define BSM_STRIDE 40     // 32 data halves + 8 pad halves -> 80B row, conflict-free

typedef _Float16 half_t;
typedef __attribute__((ext_vector_type(8)))  half_t v8h;
typedef __attribute__((ext_vector_type(16))) half_t v16h;
typedef __attribute__((ext_vector_type(8)))  float  v8f;
typedef int i32x4 __attribute__((vector_size(16)));

union HFrag { v16h v; half_t h[16]; };
union FAcc  { v8f  v; float  f[8];  };

#if __has_builtin(__builtin_amdgcn_global_load_async_to_lds_b128)
#define HAS_ASYNC_LDS 1
typedef __attribute__((address_space(1))) i32x4* gv4_p;   // global int4*
typedef __attribute__((address_space(3))) i32x4* lv4_p;   // LDS int4*
#else
#define HAS_ASYNC_LDS 0
#endif

__device__ __forceinline__ void wait_async_lds() {
#if HAS_ASYNC_LDS
#if __has_builtin(__builtin_amdgcn_s_wait_asynccnt)
    __builtin_amdgcn_s_wait_asynccnt(0);
#else
    asm volatile("s_wait_asynccnt 0" ::: "memory");
#endif
#endif
}

// Stage one 64(N) x 32(K) f16 B-chunk into LDS (row stride BSM_STRIDE halves).
// 256 threads: thread t copies 16B: row t/4, quarter t%4.
__device__ __forceinline__ void stage_b(const half_t* __restrict__ Bp,
                                        half_t* dstbase, int nbase, int k0, int tid) {
    int nl   = tid >> 2;
    int part = tid & 3;
    const half_t* gsrc = Bp + (size_t)(nbase + nl) * KDS + k0 + part * 8;
    half_t*       dst  = dstbase + nl * BSM_STRIDE + part * 8;
#if HAS_ASYNC_LDS
    __builtin_amdgcn_global_load_async_to_lds_b128((gv4_p)gsrc, (lv4_p)dst, 0, 0);
#else
    *(v8h*)dst = *(const v8h*)gsrc;
#endif
}

// ---------------- stage 1: spatial mean pool over local half ----------------
__global__ void pool_kernel(const float* __restrict__ x, float* __restrict__ pooled) {
    __shared__ float red[256];
    int bc = blockIdx.x;                  // 0..B_*MID-1
    int b = bc / MID, c = bc % MID;
    const float* p = x + ((size_t)(b * CIN2 + c)) * HW;
    float s = 0.f;
    for (int i = threadIdx.x; i < HW; i += 256) s += p[i];
    red[threadIdx.x] = s;
    __syncthreads();
    for (int off = 128; off > 0; off >>= 1) {
        if ((int)threadIdx.x < off) red[threadIdx.x] += red[threadIdx.x + off];
        __syncthreads();
    }
    if (threadIdx.x == 0) pooled[bc] = red[0] * (1.0f / (float)HW);
}

// ---------------- stage 2: kernel generator (relu(pooled @ Wkg^T)) ----------
__global__ void kgen_kernel(const float* __restrict__ pooled,
                            const float* __restrict__ Wkg,
                            float* __restrict__ kers) {
    int t = blockIdx.x * 256 + threadIdx.x;       // B_*MID*9 = 18432
    if (t >= B_ * MID * 9) return;
    int b = t / (MID * 9), r = t % (MID * 9);
    const float* pv = pooled + b * MID;
    const float* wr = Wkg + (size_t)r * MID;
    float s = 0.f;
    #pragma unroll 8
    for (int j = 0; j < MID; ++j) s += pv[j] * wr[j];
    kers[t] = fmaxf(s, 0.f);
}

// ------- stage 3: weight f16 repack + fused BN(scale,beta) precompute -------
// Wg16[n][k] = gpw_w[n][k]
// Bds16[o][tap][c] = ds_w[o][c][tap]
// s1/t1: pw epilogue  val = acc*s1[n] + t1[n]   (bias+BN folded)
// s2/t2: ds epilogue  val = acc*s2[n] + t2[n]
__global__ void prep_kernel(const float* __restrict__ gpw_w,
                            const float* __restrict__ ds_w,
                            const float* __restrict__ gpw_b,
                            const float* __restrict__ ds_b,
                            const float* __restrict__ bn1g, const float* __restrict__ bn1b,
                            const float* __restrict__ bn1m, const float* __restrict__ bn1v,
                            const float* __restrict__ bn2g, const float* __restrict__ bn2b,
                            const float* __restrict__ bn2m, const float* __restrict__ bn2v,
                            half_t* __restrict__ Wg16, half_t* __restrict__ Bds16,
                            float* __restrict__ s1, float* __restrict__ t1,
                            float* __restrict__ s2, float* __restrict__ t2) {
    int t = blockIdx.x * 256 + threadIdx.x;
    if (t < MID * MID) Wg16[t] = (half_t)gpw_w[t];
    if (t < OUTC * KDS) {
        int o   = t / KDS;
        int r   = t % KDS;
        int tap = r / CIN2;
        int c   = r % CIN2;
        Bds16[t] = (half_t)ds_w[((size_t)o * CIN2 + c) * 9 + tap];
    }
    if (t < MID) {
        float inv = bn1g[t] / sqrtf(bn1v[t] + EPSV);
        s1[t] = inv;
        t1[t] = gpw_b[t] * inv + bn1b[t] - bn1m[t] * inv;
    }
    if (t < OUTC) {
        float inv = bn2g[t] / sqrtf(bn2v[t] + EPSV);
        s2[t] = inv;
        t2[t] = ds_b[t] * inv + bn2b[t] - bn2m[t] * inv;
    }
}

// ---------------- stage 4: local per-sample depthwise 3x3, pad 1 ------------
__global__ void local_dw_kernel(const float* __restrict__ x,
                                const float* __restrict__ kers,
                                half_t* __restrict__ combined) {
    size_t t = (size_t)blockIdx.x * 256 + threadIdx.x;   // B_*MID*HW
    int xx = (int)(t % W_);
    int yy = (int)((t / W_) % H_);
    int c  = (int)((t / HW) % MID);
    int b  = (int)(t / ((size_t)HW * MID));
    const float* in = x + ((size_t)(b * CIN2 + c)) * HW;
    const float* k  = kers + ((size_t)b * MID + c) * 9;
    float s = 0.f;
    #pragma unroll
    for (int ky = 0; ky < 3; ++ky) {
        int iy = yy + ky - 1;
        if (iy < 0 || iy >= H_) continue;
        #pragma unroll
        for (int kx = 0; kx < 3; ++kx) {
            int ix = xx + kx - 1;
            if (ix < 0 || ix >= W_) continue;
            s += in[iy * W_ + ix] * k[ky * 3 + kx];
        }
    }
    combined[((size_t)b * HW + (size_t)yy * W_ + xx) * CIN2 + c] = (half_t)s;
}

// ---------------- stage 5: global depthwise 3x3 dil=2 pad=2 + bias ----------
__global__ void global_dw_kernel(const float* __restrict__ x,
                                 const float* __restrict__ gdw_w,
                                 const float* __restrict__ gdw_b,
                                 half_t* __restrict__ g1) {
    size_t t = (size_t)blockIdx.x * 256 + threadIdx.x;   // B_*MID*HW
    int xx = (int)(t % W_);
    int yy = (int)((t / W_) % H_);
    int c  = (int)((t / HW) % MID);
    int b  = (int)(t / ((size_t)HW * MID));
    const float* in = x + ((size_t)(b * CIN2 + MID + c)) * HW;
    const float* k  = gdw_w + (size_t)c * 9;
    float s = gdw_b[c];
    #pragma unroll
    for (int ky = 0; ky < 3; ++ky) {
        int iy = yy + 2 * (ky - 1);
        if (iy < 0 || iy >= H_) continue;
        #pragma unroll
        for (int kx = 0; kx < 3; ++kx) {
            int ix = xx + 2 * (kx - 1);
            if (ix < 0 || ix >= W_) continue;
            s += in[iy * W_ + ix] * k[ky * 3 + kx];
        }
    }
    g1[((size_t)b * HW + (size_t)yy * W_ + xx) * MID + c] = (half_t)s;
}

// ---------------- stage 6: 1x1 conv GEMM via WMMA (16M x 128N per wave) -----
// M = B_*HW, N = MID, K = MID. A = g1 (row-major MxK f16), Bp = Wg16 [n][k].
// Each wave: one M-tile, all 8 N-tiles -> A fragment reused 8x per K-chunk.
__global__ __launch_bounds__(256)
void gemm_pw_kernel(const half_t* __restrict__ A,
                    const half_t* __restrict__ Bp,
                    const float* __restrict__ s1,
                    const float* __restrict__ t1,
                    half_t* __restrict__ combined) {
    int lane = threadIdx.x & 31;
    int wave = threadIdx.x >> 5;
    int mt   = blockIdx.x * 8 + wave;        // M/16 tiles exactly
    int m0   = mt * 16;
    int l    = lane & 15;
    int hi   = lane >> 4;

    FAcc acc[8];
    #pragma unroll
    for (int nt = 0; nt < 8; ++nt)
        #pragma unroll
        for (int i = 0; i < 8; ++i) acc[nt].f[i] = 0.f;

    const half_t* arow  = A  + (size_t)(m0 + l) * MID + hi * 8;
    const half_t* bbase = Bp + (size_t)l * MID + hi * 8;

    #pragma unroll
    for (int k0 = 0; k0 < MID; k0 += 32) {
        HFrag a;
        v8h a0 = *(const v8h*)(arow + k0);
        v8h a1 = *(const v8h*)(arow + k0 + 16);
        #pragma unroll
        for (int i = 0; i < 8; ++i) { a.h[i] = a0[i]; a.h[8 + i] = a1[i]; }
        #pragma unroll
        for (int nt = 0; nt < 8; ++nt) {
            HFrag bf;
            const half_t* bc = bbase + (size_t)nt * 16 * MID + k0;
            v8h b0 = *(const v8h*)(bc);
            v8h b1 = *(const v8h*)(bc + 16);
            #pragma unroll
            for (int i = 0; i < 8; ++i) { bf.h[i] = b0[i]; bf.h[8 + i] = b1[i]; }
            acc[nt].v = __builtin_amdgcn_wmma_f32_16x16x32_f16(
                false, a.v, false, bf.v, (short)0, acc[nt].v, false, false);
        }
    }

    #pragma unroll
    for (int nt = 0; nt < 8; ++nt) {
        int n = nt * 16 + l;
        float sc = s1[n], tb = t1[n];
        #pragma unroll
        for (int i = 0; i < 8; ++i) {
            int m = m0 + i + hi * 8;
            float v = fmaxf(acc[nt].f[i] * sc + tb, 0.f);
            combined[(size_t)m * CIN2 + MID + n] = (half_t)v;
        }
    }
}

// ------- stage 7: 3x3 stride-2 conv as implicit GEMM via WMMA ---------------
// M = B_*OHW, N = OUTC, K = 9*256.  Block: 8 waves, each a distinct M-tile,
// all sharing one 64-channel N panel. B chunks (64N x 32K) are async-staged
// into double-buffered LDS (padded rows, conflict-free ds_load_b128), WMMA
// overlaps the next chunk's DMA. Epilogue: fused bias/BN/ReLU -> NCHW f32.
__global__ __launch_bounds__(256)
void gemm_ds_kernel(const half_t* __restrict__ combined,
                    const half_t* __restrict__ Bp,
                    const float* __restrict__ s2,
                    const float* __restrict__ t2,
                    float* __restrict__ out) {
    __shared__ __attribute__((aligned(16))) half_t Bsm[2][64 * BSM_STRIDE];

    int tid  = threadIdx.x;
    int lane = tid & 31;
    int wave = tid >> 5;
    int mg    = blockIdx.x >> 3;             // 392 M-groups (8 tiles each)
    int ng    = blockIdx.x & 7;              // 8 N-panels of 64 channels
    int m0    = (mg * 8 + wave) * 16;
    int nbase = ng * 64;
    int l  = lane & 15;
    int hi = lane >> 4;

    int m   = m0 + l;
    int b   = m / OHW;
    int rem = m % OHW;
    int oy  = rem / OW;
    int ox  = rem % OW;

    FAcc acc[4];
    #pragma unroll
    for (int nt = 0; nt < 4; ++nt)
        #pragma unroll
        for (int i = 0; i < 8; ++i) acc[nt].f[i] = 0.f;

    stage_b(Bp, &Bsm[0][0], nbase, 0, tid);

    for (int kc = 0; kc < KDS / 32; ++kc) {  // 72 iterations
        wait_async_lds();
        __syncthreads();
        int cur = kc & 1;
        if (kc + 1 < KDS / 32)
            stage_b(Bp, &Bsm[cur ^ 1][0], nbase, (kc + 1) * 32, tid);

        int k0  = kc * 32;
        int tap = k0 >> 8;                   // 256 channels per tap, 256%32==0
        int kch = k0 & 255;
        int ky  = tap / 3, kx = tap % 3;
        int iy  = 2 * oy - 1 + ky;
        int ix  = 2 * ox - 1 + kx;

        HFrag a;
        if (iy >= 0 && iy < H_ && ix >= 0 && ix < W_) {
            const half_t* arow = combined
                + ((size_t)((b * H_ + iy) * W_ + ix)) * CIN2 + kch + hi * 8;
            v8h a0 = *(const v8h*)(arow);
            v8h a1 = *(const v8h*)(arow + 16);
            #pragma unroll
            for (int i = 0; i < 8; ++i) { a.h[i] = a0[i]; a.h[8 + i] = a1[i]; }
        } else {
            #pragma unroll
            for (int i = 0; i < 16; ++i) a.h[i] = (half_t)0.f;
        }

        #pragma unroll
        for (int nt = 0; nt < 4; ++nt) {
            HFrag bf;
            const half_t* lp = &Bsm[cur][(nt * 16 + l) * BSM_STRIDE + hi * 8];
            v8h b0 = *(const v8h*)(lp);
            v8h b1 = *(const v8h*)(lp + 16);
            #pragma unroll
            for (int i = 0; i < 8; ++i) { bf.h[i] = b0[i]; bf.h[8 + i] = b1[i]; }
            acc[nt].v = __builtin_amdgcn_wmma_f32_16x16x32_f16(
                false, a.v, false, bf.v, (short)0, acc[nt].v, false, false);
        }
    }

    #pragma unroll
    for (int nt = 0; nt < 4; ++nt) {
        int n = nbase + nt * 16 + l;
        float sc = s2[n], tb = t2[n];
        #pragma unroll
        for (int i = 0; i < 8; ++i) {
            int mi = m0 + i + hi * 8;
            int b2 = mi / OHW;
            int r2 = mi % OHW;
            float v = fmaxf(acc[nt].f[i] * sc + tb, 0.f);
            out[((size_t)b2 * OUTC + n) * OHW + r2] = v;
        }
    }
}

// ---------------------------------------------------------------------------
extern "C" void kernel_launch(void* const* d_in, const int* in_sizes, int n_in,
                              void* d_out, int out_size, void* d_ws, size_t ws_size,
                              hipStream_t stream) {
    const float* x     = (const float*)d_in[0];
    const float* Wkg   = (const float*)d_in[1];
    const float* gdw_w = (const float*)d_in[2];
    const float* gdw_b = (const float*)d_in[3];
    const float* gpw_w = (const float*)d_in[4];
    const float* gpw_b = (const float*)d_in[5];
    const float* bn1g  = (const float*)d_in[6];
    const float* bn1b  = (const float*)d_in[7];
    const float* bn1m  = (const float*)d_in[8];
    const float* bn1v  = (const float*)d_in[9];
    const float* ds_w  = (const float*)d_in[10];
    const float* ds_b  = (const float*)d_in[11];
    const float* bn2g  = (const float*)d_in[12];
    const float* bn2b  = (const float*)d_in[13];
    const float* bn2m  = (const float*)d_in[14];
    const float* bn2v  = (const float*)d_in[15];
    float* out = (float*)d_out;

    // Workspace layout (all offsets 256B-aligned):
    char* ws = (char*)d_ws;
    float*  pooled   = (float*)(ws + 0);          //  2048 f32
    float*  kers     = (float*)(ws + 8192);       // 18432 f32
    float*  s1       = (float*)(ws + 81920);      //   128 f32
    float*  t1       = (float*)(ws + 82432);      //   128 f32
    float*  s2       = (float*)(ws + 82944);      //   512 f32
    float*  t2       = (float*)(ws + 84992);      //   512 f32
    half_t* Wg16     = (half_t*)(ws + 87040);     // 128*128 f16
    half_t* Bds16    = (half_t*)(ws + 119808);    // 512*2304 f16
    half_t* g1       = (half_t*)(ws + 2479104);   // B*HW*128 f16  (51.4 MB)
    half_t* combined = (half_t*)(ws + 53859328);  // B*HW*256 f16 (102.8 MB)

    pool_kernel<<<B_ * MID, 256, 0, stream>>>(x, pooled);
    kgen_kernel<<<(B_ * MID * 9 + 255) / 256, 256, 0, stream>>>(pooled, Wkg, kers);
    prep_kernel<<<(OUTC * KDS + 255) / 256, 256, 0, stream>>>(
        gpw_w, ds_w, gpw_b, ds_b,
        bn1g, bn1b, bn1m, bn1v, bn2g, bn2b, bn2m, bn2v,
        Wg16, Bds16, s1, t1, s2, t2);

    int dw_blocks = (int)(((size_t)B_ * MID * HW) / 256);      // 100352
    local_dw_kernel<<<dw_blocks, 256, 0, stream>>>(x, kers, combined);
    global_dw_kernel<<<dw_blocks, 256, 0, stream>>>(x, gdw_w, gdw_b, g1);

    int pw_blocks = (B_ * HW / 16) / 8;                        // 1568
    gemm_pw_kernel<<<pw_blocks, 256, 0, stream>>>(g1, Wg16, s1, t1, combined);

    int ds_blocks = (B_ * OHW / 16 / 8) * (OUTC / 64);         // 392*8 = 3136
    gemm_ds_kernel<<<ds_blocks, 256, 0, stream>>>(combined, Bds16, s2, t2, out);
}
